// Attention_18872086298940
// MI455X (gfx1250) — matrix operand order repaired and checked by reading, not compiled, not run on hardware
//
#include <hip/hip_runtime.h>

typedef __attribute__((ext_vector_type(16))) _Float16 v16h;
typedef __attribute__((ext_vector_type(8)))  _Float16 v8h;
typedef __attribute__((ext_vector_type(8)))  float    v8f;

#define SEQ    2048
#define DMODEL 512
#define HEADS  8
#define HD     64

// ---- CDNA5 async global->LDS copy (16B per lane) + ASYNCcnt waits ---------
__device__ __forceinline__ void async_ld16(unsigned ldsOff, const _Float16* g) {
  asm volatile("global_load_async_to_lds_b128 %0, %1, off"
               :: "v"(ldsOff), "v"(g) : "memory");
}
__device__ __forceinline__ void wait_async_le8() {
  asm volatile("s_wait_asynccnt 0x8" ::: "memory");
}
__device__ __forceinline__ void wait_async_le0() {
  asm volatile("s_wait_asynccnt 0x0" ::: "memory");
}

// ---------------------------------------------------------------------------
// Pack x: f32 (8192 x 512) -> f16 (8192 x 512), row-major. 8 elems/thread.
// ---------------------------------------------------------------------------
__global__ __launch_bounds__(256) void pack_x_kernel(
    const float* __restrict__ X, _Float16* __restrict__ Xh) {
  const size_t base = ((size_t)blockIdx.x * 256 + threadIdx.x) * 8;
  float4 f0 = *(const float4*)(X + base);
  float4 f1 = *(const float4*)(X + base + 4);
  v8h h;
  h[0]=(_Float16)f0.x; h[1]=(_Float16)f0.y; h[2]=(_Float16)f0.z; h[3]=(_Float16)f0.w;
  h[4]=(_Float16)f1.x; h[5]=(_Float16)f1.y; h[6]=(_Float16)f1.z; h[7]=(_Float16)f1.w;
  *(v8h*)(Xh + base) = h;
}

// ---------------------------------------------------------------------------
// Pack W: f32 (512 x N) row-major -> f16 transposed (N x 512).
// ---------------------------------------------------------------------------
__global__ __launch_bounds__(256) void pack_wT_kernel(
    const float* __restrict__ W, _Float16* __restrict__ WT, int N) {
  const int t  = blockIdx.x * 256 + threadIdx.x;
  const int nc = t >> 6;
  const int kc = (t & 63) * 8;
  v8h h;
  #pragma unroll
  for (int e = 0; e < 8; ++e) h[e] = (_Float16)W[(size_t)(kc + e) * N + nc];
  *(v8h*)(WT + (size_t)nc * DMODEL + kc) = h;
}

// ---------------------------------------------------------------------------
// Projection: out = Xh @ WT^T (+ bias). Wave tile 16x64, B-frags preloaded.
// ---------------------------------------------------------------------------
template <bool TRANSPOSED>
__global__ __launch_bounds__(128) void proj_kernel(
    const _Float16* __restrict__ Xh, const _Float16* __restrict__ WT,
    const float* __restrict__ bias, _Float16* __restrict__ out, int N) {
  const int lane = threadIdx.x & 31;
  const int wave = threadIdx.x >> 5;
  const int n    = lane & 15;
  const int hs   = lane >> 4;

  const int N64   = N >> 6;
  const int tile  = blockIdx.x * 4 + wave;
  const int mtile = tile / N64;
  const int n64   = tile % N64;

  const int row = mtile * 16 + n;
  const _Float16* xrow  = Xh + (size_t)row * DMODEL;
  const _Float16* wbase = WT + (size_t)(n64 * 64 + n) * DMODEL;

  v8f acc[4] = {};
  for (int kk = 0; kk < DMODEL; kk += 32) {
    v16h a;
    {
      v8h lo = *(const v8h*)(xrow + kk + hs * 16);
      v8h hi = *(const v8h*)(xrow + kk + hs * 16 + 8);
      #pragma unroll
      for (int e = 0; e < 8; ++e) { a[e] = lo[e]; a[e + 8] = hi[e]; }
    }
    v16h bf[4];
    #pragma unroll
    for (int t = 0; t < 4; ++t)
      bf[t] = *(const v16h*)(wbase + (size_t)t * 16 * DMODEL + kk + hs * 16);
    #pragma unroll
    for (int t = 0; t < 4; ++t)
      acc[t] = __builtin_amdgcn_wmma_f32_16x16x32_f16(false, a, false, bf[t],
                                                      (short)0, acc[t],
                                                      false, false);
  }

  #pragma unroll
  for (int t = 0; t < 4; ++t) {
    const int col  = n64 * 64 + t * 16 + n;
    const float bc = bias ? bias[col] : 0.0f;
    const int h    = col >> 6;
    const int sub  = col & 63;
    #pragma unroll
    for (int r = 0; r < 8; ++r) {
      const int grow = mtile * 16 + r + 8 * hs;
      const int b = grow >> 11;
      const int s = grow & (SEQ - 1);
      const float v = acc[t][r] + bc;
      if (!TRANSPOSED)
        out[(((size_t)(b * N64 + h) * SEQ) + s) * HD + sub] = (_Float16)v;
      else
        out[(((size_t)(b * HEADS + h) * HD + sub) * SEQ) + s] = (_Float16)v;
    }
  }
}

// ---------------------------------------------------------------------------
// Flash attention, cooperative + async double-buffered K/V staging.
// Block = 4 waves = 64 query rows of one (b,h). Key chunks of 64.
// Chunks ci < qb are causally full -> no masking; only the diagonal chunk
// (ci == qb) applies the causal mask.  Softmax scale folded into Q.
// LDS bytes: K0@0, K1@9216, V0@18432, V1@27648, P@36864 (+wave*2304)
//   K/V/P rows padded to 72 halves.
// ---------------------------------------------------------------------------
__global__ __launch_bounds__(128) void attn_kernel(
    const _Float16* __restrict__ Q, const _Float16* __restrict__ K,
    const _Float16* __restrict__ Vt, float* __restrict__ O) {
  __shared__ __align__(16) _Float16 smem[23040];   // 46,080 bytes

  const int tid  = threadIdx.x;
  const int lane = tid & 31;
  const int wave = tid >> 5;
  const int n    = lane & 15;
  const int hs   = lane >> 4;

  const int blk = blockIdx.x;       // B*H*(S/64) = 1024
  const int qb  = blk & 31;
  const int bh  = blk >> 5;
  const int b   = bh >> 3;
  const int h   = bh & 7;
  const int qbase = qb * 64 + wave * 16;

  const _Float16* kbb = K  + (size_t)b  * SEQ * HD;
  const _Float16* vbb = Vt + (size_t)bh * HD * SEQ;

  const unsigned smemBase = (unsigned)(size_t)smem;   // LDS byte offset
  _Float16* ldsw = smem + 18432 + wave * 1152;        // per-wave P (16x72)

  // ---- stage one 64-key chunk (K: 64x64, V: 64x64; rows padded to 72)
  auto stage = [&](int jj, int buf) {
    #pragma unroll
    for (int i = 0; i < 4; ++i) {
      const int cid = tid + i * 128;      // 0..511 16B chunks
      const int row = cid >> 3;           // 0..63
      const int c8  = cid & 7;
      const unsigned loff =
          (unsigned)(buf * 9216) + (unsigned)((row * 72 + c8 * 8) * 2);
      async_ld16(smemBase + loff,          kbb + (size_t)(jj + row) * HD + c8 * 8);
      async_ld16(smemBase + 18432u + loff, vbb + (size_t)row * SEQ + jj + c8 * 8);
    }
  };

  // ---- Q A-fragments, softmax scale folded in
  const float scale = 0.02209708691207961f;   // 1/sqrt(2048)
  v16h qa[2];
  {
    const _Float16* qrow = Q + ((size_t)bh * SEQ + (qbase + n)) * HD;
    #pragma unroll
    for (int c = 0; c < 2; ++c) {
      v8h lo = *(const v8h*)(qrow + c * 32 + hs * 16);
      v8h hi = *(const v8h*)(qrow + c * 32 + hs * 16 + 8);
      #pragma unroll
      for (int e = 0; e < 8; ++e) {
        qa[c][e]     = (_Float16)((float)lo[e] * scale);
        qa[c][e + 8] = (_Float16)((float)hi[e] * scale);
      }
    }
  }

  v8f out4[4] = {};
  float m_r[8], l_r[8];
  #pragma unroll
  for (int r = 0; r < 8; ++r) { m_r[r] = -1e30f; l_r[r] = 0.0f; }

  // ---- one chunk of work (masked selects fold away when masked==false)
  auto chunk_body = [&](int jj, int cur, bool masked) {
    const _Float16* Kc = smem + cur * 4608;          // 64 x 72 halves
    const _Float16* Vc = smem + 9216 + cur * 4608;   // 64 x 72 halves

    v16h kf0[4], kf1[4];
    #pragma unroll
    for (int t = 0; t < 4; ++t) {
      const _Float16* kp = Kc + (t * 16 + n) * 72 + hs * 16;
      kf0[t] = *(const v16h*)(kp);
      kf1[t] = *(const v16h*)(kp + 32);
    }
    v8f sc[4];
    #pragma unroll
    for (int t = 0; t < 4; ++t) {
      v8f s = {};
      s = __builtin_amdgcn_wmma_f32_16x16x32_f16(false, qa[0], false, kf0[t],
                                                 (short)0, s, false, false);
      s = __builtin_amdgcn_wmma_f32_16x16x32_f16(false, qa[1], false, kf1[t],
                                                 (short)0, s, false, false);
      sc[t] = s;
    }

    float corr[8];
    #pragma unroll
    for (int r = 0; r < 8; ++r) {
      const int qrow = qbase + r + 8 * hs;
      float sv[4];
      #pragma unroll
      for (int t = 0; t < 4; ++t)
        sv[t] = (!masked || (jj + t * 16 + n <= qrow)) ? sc[t][r] : -1e30f;
      float mx = fmaxf(fmaxf(sv[0], sv[1]), fmaxf(sv[2], sv[3]));
      #pragma unroll
      for (int off = 8; off >= 1; off >>= 1)
        mx = fmaxf(mx, __shfl_xor(mx, off, 32));
      const float mn = fmaxf(m_r[r], mx);
      const float c0 = __expf(m_r[r] - mn);
      m_r[r] = mn;
      float sum = 0.0f;
      _Float16* prow = ldsw + (r + 8 * hs) * 72;
      #pragma unroll
      for (int t = 0; t < 4; ++t) {
        const float p = __expf(sv[t] - mn);
        sum += p;
        prow[t * 16 + n] = (_Float16)p;
      }
      #pragma unroll
      for (int off = 8; off >= 1; off >>= 1)
        sum += __shfl_xor(sum, off, 32);
      l_r[r] = l_r[r] * c0 + sum;
      corr[r] = c0;
    }
    #pragma unroll
    for (int t = 0; t < 4; ++t)
      #pragma unroll
      for (int r = 0; r < 8; ++r) out4[t][r] *= corr[r];

    __builtin_amdgcn_wave_barrier();

    v16h pf[2];
    {
      const _Float16* prow = ldsw + n * 72;
      #pragma unroll
      for (int c = 0; c < 2; ++c) {
        v8h lo = *(const v8h*)(prow + c * 32 + hs * 16);
        v8h hi = *(const v8h*)(prow + c * 32 + hs * 16 + 8);
        #pragma unroll
        for (int e = 0; e < 8; ++e) { pf[c][e] = lo[e]; pf[c][e + 8] = hi[e]; }
      }
    }
    __builtin_amdgcn_wave_barrier();

    #pragma unroll
    for (int t = 0; t < 4; ++t) {
      const _Float16* vp = Vc + (t * 16 + n) * 72 + hs * 16;
      v16h vf0 = *(const v16h*)(vp);
      v16h vf1 = *(const v16h*)(vp + 32);
      out4[t] = __builtin_amdgcn_wmma_f32_16x16x32_f16(false, pf[0], false, vf0,
                                                       (short)0, out4[t],
                                                       false, false);
      out4[t] = __builtin_amdgcn_wmma_f32_16x16x32_f16(false, pf[1], false, vf1,
                                                       (short)0, out4[t],
                                                       false, false);
    }
  };

  // ---- prologue: stage chunk 0 into buffer 0
  stage(0, 0);

  // ---- full (unmasked) chunks, double-buffered
  for (int ci = 0; ci < qb; ++ci) {
    stage((ci + 1) * 64, (ci + 1) & 1);
    wait_async_le8();
    __syncthreads();
    chunk_body(ci * 64, ci & 1, false);
    __syncthreads();
  }

  // ---- diagonal chunk (the only one needing the causal mask)
  wait_async_le0();
  __syncthreads();
  chunk_body(qb * 64, qb & 1, true);

  // ---- normalize and store f32 output
  #pragma unroll
  for (int t = 0; t < 4; ++t)
    #pragma unroll
    for (int r = 0; r < 8; ++r) {
      const int qrow = qbase + r + 8 * hs;
      O[((size_t)b * SEQ + qrow) * (HEADS * HD) + h * HD + t * 16 + n] =
          out4[t][r] / l_r[r];
    }
}

// ---------------------------------------------------------------------------
extern "C" void kernel_launch(void* const* d_in, const int* in_sizes, int n_in,
                              void* d_out, int out_size, void* d_ws, size_t ws_size,
                              hipStream_t stream) {
  const float* x  = (const float*)d_in[0];
  const float* Wq = (const float*)d_in[1];
  const float* bq = (const float*)d_in[2];
  const float* Wk = (const float*)d_in[3];
  const float* Wv = (const float*)d_in[4];
  const float* bv = (const float*)d_in[5];

  _Float16* ws  = (_Float16*)d_ws;
  _Float16* qh  = ws;                        // (4,8,2048,64)  4,194,304 halves
  _Float16* kws = qh  + 4194304;             // (4,2048,64)      524,288
  _Float16* vT  = kws + 524288;              // (4,8,64,2048)  4,194,304
  _Float16* xh  = vT  + 4194304;             // (8192,512)     4,194,304
  _Float16* wqT = xh  + 4194304;             // (512,512)        262,144
  _Float16* wkT = wqT + 262144;              // (64,512)          32,768
  _Float16* wvT = wkT + 32768;               // (512,512)        262,144

  pack_x_kernel <<<2048, 256, 0, stream>>>(x, xh);
  pack_wT_kernel<<<128,  256, 0, stream>>>(Wq, wqT, 512);
  pack_wT_kernel<<<16,   256, 0, stream>>>(Wk, wkT, 64);
  pack_wT_kernel<<<128,  256, 0, stream>>>(Wv, wvT, 512);

  proj_kernel<false><<<1024, 128, 0, stream>>>(xh, wqT, bq,      qh,  512);
  proj_kernel<false><<<128,  128, 0, stream>>>(xh, wkT, nullptr, kws, 64);
  proj_kernel<true> <<<1024, 128, 0, stream>>>(xh, wvT, bv,      vT,  512);

  attn_kernel<<<1024, 128, 0, stream>>>(qh, kws, vT, (float*)d_out);
}
